// GQASelfAttention_63763084476554
// MI455X (gfx1250) — matrix operand order repaired and checked by reading, not compiled, not run on hardware
//
#include <hip/hip_runtime.h>
#include <hip/hip_bf16.h>

// ---------------------------------------------------------------------------
// GQA self-attention for MI455X (gfx1250, wave32, WMMA).
// B=4, S=2048, HID=2048, H=16, KV=4, D=128, REP=4.
// f16 WMMA (f32 accum) for all GEMMs; f32 scalar math for norm/rope/softmax.
// K tile staged via async global->LDS (ASYNCcnt); V tile VALU-transposed.
// ---------------------------------------------------------------------------

#define B_   4
#define S_   2048
#define HID_ 2048
#define H_   16
#define KV_  4
#define D_   128
#define REP_ 4
#define EPS_ 1e-6f

typedef __attribute__((ext_vector_type(16))) _Float16 v16h;
typedef __attribute__((ext_vector_type(8)))  float    v8f;

union Frag32 {            // one WMMA 16-bit A/B operand: 16 halves = 32 bytes
    uint4 u4[2];
    v16h  h16;
};
union H8 {                // 8 halves = 16 bytes
    uint4    u;
    _Float16 h[8];
};

__device__ __forceinline__ float hmax16(float v) {   // reduce within 16-lane half
    v = fmaxf(v, __shfl_xor(v, 1));
    v = fmaxf(v, __shfl_xor(v, 2));
    v = fmaxf(v, __shfl_xor(v, 4));
    v = fmaxf(v, __shfl_xor(v, 8));
    return v;
}
__device__ __forceinline__ float hsum16(float v) {
    v += __shfl_xor(v, 1);
    v += __shfl_xor(v, 2);
    v += __shfl_xor(v, 4);
    v += __shfl_xor(v, 8);
    return v;
}

// async global->LDS copy of 16 bytes (per lane); tracked by ASYNCcnt
__device__ __forceinline__ void async_copy_b128(_Float16* lds_dst,
                                                const _Float16* gaddr) {
    unsigned lds_off = (unsigned)(uintptr_t)lds_dst;   // addr[31:0] = LDS offset
    asm volatile("global_load_async_to_lds_b128 %0, %1, off"
                 :
                 : "v"(lds_off), "v"(gaddr)
                 : "memory");
}
__device__ __forceinline__ void wait_asynccnt0() {
    asm volatile("s_wait_asynccnt 0x0" ::: "memory");
}

// ---------------------------------------------------------------------------
// Kernel 1: f32 -> f16 convert (exact sizes, 4 elems/thread)
// ---------------------------------------------------------------------------
__global__ void cvt_f32_f16(const float* __restrict__ s, _Float16* __restrict__ d) {
    size_t i = ((size_t)blockIdx.x * blockDim.x + threadIdx.x) * 4;
    float4 v = *(const float4*)(s + i);
    H8 o;
    o.h[0] = (_Float16)v.x; o.h[1] = (_Float16)v.y;
    o.h[2] = (_Float16)v.z; o.h[3] = (_Float16)v.w;
    *(uint2*)(d + i) = *(uint2*)&o;   // 8-byte store
}

// ---------------------------------------------------------------------------
// Kernel 2/5: C[M,N] = A[M,K] * B[N,K]^T  (both K-contiguous row-major).
// Block = 256 threads = 8 waves arranged 4(M) x 2(N); wave does 16x64.
// ---------------------------------------------------------------------------
template <typename OutT>
__global__ void __launch_bounds__(256) gemm_nt(const _Float16* __restrict__ A,
                                               const _Float16* __restrict__ Bm,
                                               OutT* __restrict__ C,
                                               int M, int N, int K) {
    const int lane = threadIdx.x & 31;
    const int wave = threadIdx.x >> 5;
    const int m0 = blockIdx.y * 64 + (wave & 3) * 16;
    const int n0 = blockIdx.x * 128 + (wave >> 2) * 64;
    const int lh8  = (lane >> 4) * 8;    // 0 or 8
    const int lh16 = (lane >> 4) * 16;   // 0 or 16
    const int ln   = lane & 15;

    v8f acc[4] = {};
    for (int kb = 0; kb < K; kb += 32) {
        Frag32 a;
        const _Float16* ap = A + (size_t)(m0 + ln) * K + kb + lh8;
        a.u4[0] = *(const uint4*)ap;
        a.u4[1] = *(const uint4*)(ap + 16);
#pragma unroll
        for (int nt = 0; nt < 4; ++nt) {
            Frag32 bf;
            const _Float16* bp = Bm + (size_t)(n0 + nt * 16 + ln) * K + kb + lh8;
            bf.u4[0] = *(const uint4*)bp;
            bf.u4[1] = *(const uint4*)(bp + 16);
            acc[nt] = __builtin_amdgcn_wmma_f32_16x16x32_f16(
                false, a.h16, false, bf.h16, (short)0, acc[nt], false, false);
        }
    }
#pragma unroll
    for (int nt = 0; nt < 4; ++nt)
#pragma unroll
        for (int r = 0; r < 8; ++r) {
            int row = m0 + lh8 + r;
            int col = n0 + nt * 16 + ln;
            C[(size_t)row * N + col] = (OutT)acc[nt][r];
        }
}

// ---------------------------------------------------------------------------
// Kernel 3: per-(b,s,head-slot) RMSNorm + RoPE, write Q/K/V caches (f16).
// grid = (B*S, H+KV+KV); block = 32 (one wave, 4 elems/lane).
// ---------------------------------------------------------------------------
__global__ void __launch_bounds__(32) norm_rope(const _Float16* __restrict__ qkv,
                                                const float* __restrict__ rc,
                                                const float* __restrict__ qw,
                                                const float* __restrict__ kw,
                                                _Float16* __restrict__ Qc,
                                                _Float16* __restrict__ Kc,
                                                _Float16* __restrict__ Vc) {
    const int bs = blockIdx.x;
    const int hs = blockIdx.y;            // 0..15 q, 16..19 k, 20..23 v
    const int lane = threadIdx.x;
    const int b = bs >> 11, s = bs & (S_ - 1);
    const int d0 = lane * 4;

    const _Float16* src = qkv + (size_t)bs * (H_ + 2 * KV_) * D_ + hs * D_ + d0;
    float y0 = (float)src[0], y1 = (float)src[1], y2 = (float)src[2], y3 = (float)src[3];

    _Float16* dst;
    const float* w;
    bool plain = false;
    if (hs < H_) {
        dst = Qc + ((size_t)(b * H_ + hs) * S_ + s) * D_;  w = qw;
    } else if (hs < H_ + KV_) {
        dst = Kc + ((size_t)(b * KV_ + hs - H_) * S_ + s) * D_;  w = kw;
    } else {
        dst = Vc + ((size_t)(b * KV_ + hs - H_ - KV_) * S_ + s) * D_;
        w = kw; plain = true;
    }

    if (!plain) {
        float ss = y0 * y0 + y1 * y1 + y2 * y2 + y3 * y3;
#pragma unroll
        for (int off = 16; off; off >>= 1) ss += __shfl_xor(ss, off);
        float rn = rsqrtf(ss * (1.0f / (float)D_) + EPS_);
        y0 *= rn * w[d0 + 0]; y1 *= rn * w[d0 + 1];
        y2 *= rn * w[d0 + 2]; y3 *= rn * w[d0 + 3];
        // rope: pairs (y0,y1) idx i=2*lane, (y2,y3) idx 2*lane+1; rc (S, D/2, 2)
        const float* rcs = rc + (size_t)s * D_;
        float c0 = rcs[(2 * lane) * 2],     s0 = rcs[(2 * lane) * 2 + 1];
        float c1 = rcs[(2 * lane + 1) * 2], s1 = rcs[(2 * lane + 1) * 2 + 1];
        float o0 = y0 * c0 - y1 * s0, o1 = y0 * s0 + y1 * c0;
        float o2 = y2 * c1 - y3 * s1, o3 = y2 * s1 + y3 * c1;
        y0 = o0; y1 = o1; y2 = o2; y3 = o3;
    }
    H8 o;
    o.h[0] = (_Float16)y0; o.h[1] = (_Float16)y1;
    o.h[2] = (_Float16)y2; o.h[3] = (_Float16)y3;
    *(uint2*)(dst + d0) = *(uint2*)&o;
}

// ---------------------------------------------------------------------------
// Kernel 4: flash attention. grid = (S/128, H, B); block = 256 (8 waves).
// Wave owns 16 query rows; loop over 32-key tiles staged in LDS.
// K tile: async global->LDS copy. V tile: VALU transpose into LDS.
// ---------------------------------------------------------------------------
#define KSTRIDE 136   // LDS K-tile row stride (halves): bank-conflict free
#define VSTRIDE 40    // LDS V^T / P row stride (halves)

__global__ void __launch_bounds__(256) attn(const _Float16* __restrict__ Qc,
                                            const _Float16* __restrict__ Kc,
                                            const _Float16* __restrict__ Vc,
                                            _Float16* __restrict__ O) {
    const int lane = threadIdx.x & 31;
    const int wave = threadIdx.x >> 5;
    const int h = blockIdx.y, b = blockIdx.z, kvh = h >> 2;
    const int q0 = blockIdx.x * 128 + wave * 16;
    const int lh8  = (lane >> 4) * 8;
    const int lh16 = (lane >> 4) * 16;
    const int ln   = lane & 15;

    __shared__ _Float16 sK[32 * KSTRIDE];        // [key][d]
    __shared__ _Float16 sVT[D_ * VSTRIDE];       // [d][key]
    __shared__ _Float16 sP[8][16 * VSTRIDE];     // per-wave [row][key]

    const _Float16* Qg = Qc + (size_t)(b * H_ + h) * S_ * D_;
    const _Float16* Kg = Kc + (size_t)(b * KV_ + kvh) * S_ * D_;
    const _Float16* Vg = Vc + (size_t)(b * KV_ + kvh) * S_ * D_;

    // Q fragments (A operand), 4 d-chunks of 32, resident all loop
    Frag32 qa[4];
    {
        const _Float16* qp = Qg + (size_t)(q0 + ln) * D_ + lh8;
#pragma unroll
        for (int c = 0; c < 4; ++c) {
            qa[c].u4[0] = *(const uint4*)(qp + c * 32);
            qa[c].u4[1] = *(const uint4*)(qp + c * 32 + 16);
        }
    }

    v8f o[8] = {};
    float m[8], l[8];
#pragma unroll
    for (int r = 0; r < 8; ++r) { m[r] = -INFINITY; l[r] = 0.0f; }
    const float scale = 0.088388347648318447f;   // 1/sqrt(128)

    for (int kt = 0; kt < S_ / 32; ++kt) {
        __syncthreads();   // all waves done reading previous sK/sVT
        {   // K tile: 32 keys x 128 d, async DMA global->LDS (2x b128/thread)
            int key = threadIdx.x >> 3, seg = threadIdx.x & 7;
            const _Float16* kp = Kg + (size_t)(kt * 32 + key) * D_ + seg * 16;
            _Float16* kd = &sK[key * KSTRIDE + seg * 16];
            async_copy_b128(kd, kp);
            async_copy_b128(kd + 8, kp + 8);
            if (kt + 1 < S_ / 32) __builtin_prefetch(kp + 32 * D_, 0, 0);
        }
        {   // V^T tile: sVT[d][key], packed 2 keys/b32 (VALU transpose)
            int kp2 = threadIdx.x & 15, dseg = threadIdx.x >> 4;
            const _Float16* vp = Vg + (size_t)(kt * 32 + kp2 * 2) * D_ + dseg * 8;
            H8 va, vb;
            va.u = *(const uint4*)vp;
            vb.u = *(const uint4*)(vp + D_);
#pragma unroll
            for (int i = 0; i < 8; ++i) {
                int d = dseg * 8 + i;
                sVT[d * VSTRIDE + kp2 * 2]     = va.h[i];
                sVT[d * VSTRIDE + kp2 * 2 + 1] = vb.h[i];
            }
        }
        wait_asynccnt0();  // this wave's async K-tile writes landed in LDS
        __syncthreads();   // tile visible to all waves

        // scores: 16 queries x 32 keys (two 16-key subtiles)
        v8f s0 = {}, s1 = {};
#pragma unroll
        for (int c = 0; c < 4; ++c) {
            Frag32 k0, k1;
            int b0 = ln * KSTRIDE + c * 32 + lh8;
            int b1 = (ln + 16) * KSTRIDE + c * 32 + lh8;
            k0.u4[0] = *(const uint4*)&sK[b0]; k0.u4[1] = *(const uint4*)&sK[b0 + 16];
            k1.u4[0] = *(const uint4*)&sK[b1]; k1.u4[1] = *(const uint4*)&sK[b1 + 16];
            s0 = __builtin_amdgcn_wmma_f32_16x16x32_f16(false, qa[c].h16, false, k0.h16,
                                                        (short)0, s0, false, false);
            s1 = __builtin_amdgcn_wmma_f32_16x16x32_f16(false, qa[c].h16, false, k1.h16,
                                                        (short)0, s1, false, false);
        }

        // online softmax update (row r lives in C-vgpr r across one 16-lane half)
        float p0[8], p1[8];
#pragma unroll
        for (int r = 0; r < 8; ++r) {
            float v0 = s0[r] * scale, v1 = s1[r] * scale;
            float mn = fmaxf(m[r], hmax16(fmaxf(v0, v1)));
            float alpha = __expf(m[r] - mn);
            p0[r] = __expf(v0 - mn);
            p1[r] = __expf(v1 - mn);
            l[r] = l[r] * alpha + hsum16(p0[r] + p1[r]);
            m[r] = mn;
#pragma unroll
            for (int c = 0; c < 8; ++c) o[c][r] *= alpha;
        }

        // P -> LDS (wave-private), re-read as 16x32 A operand
        _Float16* Pw = sP[wave];
#pragma unroll
        for (int r = 0; r < 8; ++r) {
            int row = lh8 + r;
            Pw[row * VSTRIDE + ln]      = (_Float16)p0[r];
            Pw[row * VSTRIDE + 16 + ln] = (_Float16)p1[r];
        }
        Frag32 pa;
        {
            int base = ln * VSTRIDE + lh8;
            pa.u4[0] = *(const uint4*)&Pw[base];
            pa.u4[1] = *(const uint4*)&Pw[base + 16];
        }
        // O += P * V  (8 d-chunks of 16)
#pragma unroll
        for (int c = 0; c < 8; ++c) {
            Frag32 vb;
            int base = (c * 16 + ln) * VSTRIDE + lh8;
            vb.u4[0] = *(const uint4*)&sVT[base];
            vb.u4[1] = *(const uint4*)&sVT[base + 16];
            o[c] = __builtin_amdgcn_wmma_f32_16x16x32_f16(false, pa.h16, false, vb.h16,
                                                          (short)0, o[c], false, false);
        }
    }

    // epilogue: O[b][s][h*D+d] f16
#pragma unroll
    for (int c = 0; c < 8; ++c)
#pragma unroll
        for (int r = 0; r < 8; ++r) {
            int row = q0 + lh8 + r;
            int col = h * D_ + c * 16 + ln;
            O[((size_t)b * S_ + row) * (H_ * D_) + col] = (_Float16)(o[c][r] / l[r]);
        }
}

// ---------------------------------------------------------------------------
// Workspace layout (bytes). O aliases XH (x-f16 no longer needed by then).
// ---------------------------------------------------------------------------
static constexpr size_t XH_OFF   = 0;                       // 8192*2048*2 = 32 MB (also O)
static constexpr size_t WCAT_OFF = XH_OFF   + 33554432;     // 3072*2048*2 = 12 MB
static constexpr size_t WO_OFF   = WCAT_OFF + 12582912;     // 2048*2048*2 =  8 MB
static constexpr size_t QKV_OFF  = WO_OFF   + 8388608;      // 8192*3072*2 = 48 MB
static constexpr size_t Q_OFF    = QKV_OFF  + 50331648;     // 16 heads    = 32 MB
static constexpr size_t K_OFF    = Q_OFF    + 33554432;     //  4 heads    =  8 MB
static constexpr size_t V_OFF    = K_OFF    + 8388608;      //  4 heads    =  8 MB

extern "C" void kernel_launch(void* const* d_in, const int* in_sizes, int n_in,
                              void* d_out, int out_size, void* d_ws, size_t ws_size,
                              hipStream_t stream) {
    (void)in_sizes; (void)n_in; (void)out_size; (void)ws_size;
    const float* x  = (const float*)d_in[0];
    const float* rc = (const float*)d_in[1];
    const float* Wq = (const float*)d_in[2];
    const float* Wk = (const float*)d_in[3];
    const float* Wv = (const float*)d_in[4];
    const float* Wo = (const float*)d_in[5];
    const float* qw = (const float*)d_in[6];
    const float* kw = (const float*)d_in[7];

    char* ws = (char*)d_ws;
    _Float16* xh   = (_Float16*)(ws + XH_OFF);
    _Float16* wcat = (_Float16*)(ws + WCAT_OFF);
    _Float16* woh  = (_Float16*)(ws + WO_OFF);
    _Float16* qkv  = (_Float16*)(ws + QKV_OFF);
    _Float16* Qc   = (_Float16*)(ws + Q_OFF);
    _Float16* Kc   = (_Float16*)(ws + K_OFF);
    _Float16* Vc   = (_Float16*)(ws + V_OFF);
    _Float16* Ob   = (_Float16*)(ws + XH_OFF);   // alias over xh
    float*    out  = (float*)d_out;

    // 1) f32 -> f16 converts
    cvt_f32_f16<<<(B_ * S_ * HID_) / 1024, 256, 0, stream>>>(x, xh);
    cvt_f32_f16<<<(H_ * D_ * HID_) / 1024, 256, 0, stream>>>(Wq, wcat);
    cvt_f32_f16<<<(KV_ * D_ * HID_) / 1024, 256, 0, stream>>>(Wk, wcat + (size_t)H_ * D_ * HID_);
    cvt_f32_f16<<<(KV_ * D_ * HID_) / 1024, 256, 0, stream>>>(Wv, wcat + (size_t)(H_ + KV_) * D_ * HID_);
    cvt_f32_f16<<<(HID_ * H_ * D_) / 1024, 256, 0, stream>>>(Wo, woh);

    // 2) fused QKV projection: [8192 x 3072] = xh[8192 x 2048] * Wcat^T
    gemm_nt<_Float16><<<dim3((H_ + 2 * KV_) * D_ / 128, B_ * S_ / 64), 256, 0, stream>>>(
        xh, wcat, qkv, B_ * S_, (H_ + 2 * KV_) * D_, HID_);

    // 3) RMSNorm + RoPE -> Q/K/V caches [b][head][s][d]
    norm_rope<<<dim3(B_ * S_, H_ + 2 * KV_), 32, 0, stream>>>(qkv, rc, qw, kw, Qc, Kc, Vc);

    // 4) flash attention -> Ob [b][s][h*D+d]
    attn<<<dim3(S_ / 128, H_, B_), 256, 0, stream>>>(Qc, Kc, Vc, Ob);

    // 5) output projection: out[8192 x 2048] = Ob * Wo^T (f32 out)
    gemm_nt<float><<<dim3(HID_ / 128, B_ * S_ / 64), 256, 0, stream>>>(
        Ob, woh, out, B_ * S_, HID_, H_ * D_);
}
